// MessagePassingLayer_43731357008326
// MI455X (gfx1250) — matrix operand order repaired and checked by reading, not compiled, not run on hardware
//
#include <hip/hip_runtime.h>
#include <hip/hip_bf16.h>

typedef unsigned short ushort_t;
typedef __attribute__((ext_vector_type(16))) __bf16 v16bf;
typedef __attribute__((ext_vector_type(8)))  float  v8f;
typedef __attribute__((ext_vector_type(4)))  unsigned int u32x4;

#define EMB 128
#define K4D 512
#define XCOLS 768          // [Hh0 | E | Hh0*r | E*r | Hh1 | Hh1*r]
#define XPITCH 776         // +8 bf16 pad -> 1552B row stride, 388 dw %64 = 4 -> conflict-free

union FragU {
    v16bf v;
    u32x4 q[2];
    ushort_t u[16];
};

static __device__ __forceinline__ ushort_t f2bf_rn(float f) {
    unsigned u = __float_as_uint(f);
    unsigned r = u + 0x7FFFu + ((u >> 16) & 1u);
    return (ushort_t)(r >> 16);
}
static __device__ __forceinline__ float bf2f(ushort_t h) {
    return __uint_as_float(((unsigned)h) << 16);
}
static __device__ __forceinline__ void store_hl(ushort_t* rh, ushort_t* rl, int idx, float v) {
    ushort_t hi = f2bf_rn(v);
    rh[idx] = hi;
    rl[idx] = f2bf_rn(v - bf2f(hi));
}

// ---------------- zero scratch ----------------
__global__ void mp_zero_kernel(float* p, long n) {
    long i = (long)blockIdx.x * blockDim.x + threadIdx.x;
    long stride = (long)gridDim.x * blockDim.x;
    for (; i < n; i += stride) p[i] = 0.0f;
}

// ---------------- W -> bf16 hi/lo split ----------------
__global__ void mp_wprep_kernel(const float* __restrict__ Wf, const float* __restrict__ Wb,
                                ushort_t* Whi_f, ushort_t* Wlo_f,
                                ushort_t* Whi_b, ushort_t* Wlo_b) {
    int i = blockIdx.x * blockDim.x + threadIdx.x;
    if (i < EMB * K4D) {
        store_hl(Whi_f, Wlo_f, i, Wf[i]);
        store_hl(Whi_b, Wlo_b, i, Wb[i]);
    }
}

// ---------------- per-direction receive counts ----------------
__global__ void mp_counts_kernel(const int* __restrict__ ht, float* cntf, float* cntb, int m) {
    int e = blockIdx.x * blockDim.x + threadIdx.x;
    if (e < m) {
        atomicAdd(&cntf[ht[2 * e + 1]], 1.0f);  // fwd messages land on ht[:,1]
        atomicAdd(&cntb[ht[2 * e + 0]], 1.0f);  // back messages land on ht[:,0]
    }
}

// ---------------- main: fused gather + bf16x3 WMMA GEMM + atomic scatter ----------------
__global__ __launch_bounds__(128) void mp_gemm_scatter_kernel(
    const float* __restrict__ H, const float* __restrict__ E,
    const int* __restrict__ ht, const float* __restrict__ R,
    const ushort_t* __restrict__ Whi_f, const ushort_t* __restrict__ Wlo_f,
    const ushort_t* __restrict__ Whi_b, const ushort_t* __restrict__ Wlo_b,
    float* __restrict__ agg, int m)
{
    __shared__ ushort_t sHi[16 * XPITCH];
    __shared__ ushort_t sLo[16 * XPITCH];
    __shared__ int sdst[2][16];

    const int tid = threadIdx.x;      // 0..127
    const int c = tid;                // channel within embedding
    const long e0 = (long)blockIdx.x * 16;

    // ---- stage X tile (hi/lo bf16) ----
    for (int le = 0; le < 16; ++le) {
        long e = e0 + le;
        float he0 = 0.f, he1 = 0.f, ee = 0.f, rr = 0.f;
        int h0 = 0, h1 = 0;
        if (e < m) {
            h0 = ht[2 * e];
            h1 = ht[2 * e + 1];
            he0 = H[(long)h0 * EMB + c];
            he1 = H[(long)h1 * EMB + c];
            ee  = E[e * EMB + c];
            rr  = R[e * EMB + c];
        }
        if (tid == 0) {
            sdst[0][le] = (e < m) ? h1 : 0;   // fwd dst
            sdst[1][le] = (e < m) ? h0 : 0;   // back dst
        }
        ushort_t* rh = &sHi[le * XPITCH];
        ushort_t* rl = &sLo[le * XPITCH];
        store_hl(rh, rl, 0 * EMB + c, he0);
        store_hl(rh, rl, 1 * EMB + c, ee);
        store_hl(rh, rl, 2 * EMB + c, he0 * rr);
        store_hl(rh, rl, 3 * EMB + c, ee * rr);
        store_hl(rh, rl, 4 * EMB + c, he1);
        store_hl(rh, rl, 5 * EMB + c, he1 * rr);
    }
    __syncthreads();

    const int lane = tid & 31;
    const int wave = tid >> 5;
    const int halfsel = (lane >> 4) & 1;
    const int nlane = lane & 15;
    const int m16 = lane & 15;

    // 16 jobs = {dir 0..1} x {n-tile 0..7}; 4 waves round-robin
    for (int job = wave; job < 16; job += 4) {
        const int dir = job >> 3;
        const int nt  = job & 7;
        const ushort_t* __restrict__ Whi = dir ? Whi_b : Whi_f;
        const ushort_t* __restrict__ Wlo = dir ? Wlo_b : Wlo_f;
        const int n = nt * 16 + nlane;

        v8f cacc = {};
        for (int kt = 0; kt < 16; ++kt) {
            const int kb = kt * 32;
            // map logical k-range [kb,kb+32) into fused X columns for this direction
            int colb = kb;
            if (dir) colb = kb + (kb < 128 ? 512 : ((kb >= 256 && kb < 384) ? 384 : 0));

            FragU ah, al, bh, bl;
            const int aoff = m16 * XPITCH + colb + halfsel * 8;
            ah.q[0] = *(const u32x4*)&sHi[aoff];
            ah.q[1] = *(const u32x4*)&sHi[aoff + 16];
            al.q[0] = *(const u32x4*)&sLo[aoff];
            al.q[1] = *(const u32x4*)&sLo[aoff + 16];

            const int boff = n * K4D + kb + halfsel * 8;
            bh.q[0] = *(const u32x4*)&Whi[boff];
            bh.q[1] = *(const u32x4*)&Whi[boff + 16];
            bl.q[0] = *(const u32x4*)&Wlo[boff];
            bl.q[1] = *(const u32x4*)&Wlo[boff + 16];

            // fp32-accuracy emulation: hi*hi + hi*lo + lo*hi
            cacc = __builtin_amdgcn_wmma_f32_16x16x32_bf16(false, ah.v, false, bh.v,
                                                           (short)0, cacc, false, false);
            cacc = __builtin_amdgcn_wmma_f32_16x16x32_bf16(false, ah.v, false, bl.v,
                                                           (short)0, cacc, false, false);
            cacc = __builtin_amdgcn_wmma_f32_16x16x32_bf16(false, al.v, false, bh.v,
                                                           (short)0, cacc, false, false);
        }

        // scatter D tile: lane holds col n, rows r(+8 for high half)
        #pragma unroll
        for (int r = 0; r < 8; ++r) {
            const int mrow = r + halfsel * 8;
            const int node = sdst[dir][mrow];
            atomicAdd(&agg[(long)node * EMB + n], cacc[r]);
        }
    }
}

// ---------------- finalize: mean + bias + LeakyReLU + residual + LayerNorm ----------------
__global__ __launch_bounds__(256) void mp_finalize_kernel(
    float* __restrict__ out, const float* __restrict__ H,
    const float* __restrict__ cntf, const float* __restrict__ cntb,
    const float* __restrict__ bf, const float* __restrict__ bb,
    const float* __restrict__ gamma, const float* __restrict__ beta, int n_nodes)
{
    const int node = (int)((blockIdx.x * blockDim.x + threadIdx.x) >> 5);
    const int lane = threadIdx.x & 31;
    if (node >= n_nodes) return;

    const long base = (long)node * EMB;
    const float cf = cntf[node], cb = cntb[node];
    const float inv = 1.0f / (cf + cb);

    float x[4], s = 0.f, sq = 0.f;
    #pragma unroll
    for (int j = 0; j < 4; ++j) {
        const int ch = lane + j * 32;
        float a = out[base + ch];                       // raw message sum (no bias)
        a = (a + cf * bf[ch] + cb * bb[ch]) * inv;      // exact bias + mean
        a = (a >= 0.f) ? a : 0.01f * a;                 // LeakyReLU(0.01)
        a += H[base + ch];                              // residual
        x[j] = a; s += a; sq += a * a;
    }
    #pragma unroll
    for (int off = 16; off >= 1; off >>= 1) {
        s  += __shfl_xor(s,  off, 32);
        sq += __shfl_xor(sq, off, 32);
    }
    const float mean = s * (1.0f / EMB);
    const float var  = sq * (1.0f / EMB) - mean * mean;
    const float rstd = rsqrtf(var + 1e-5f);
    #pragma unroll
    for (int j = 0; j < 4; ++j) {
        const int ch = lane + j * 32;
        out[base + ch] = (x[j] - mean) * rstd * gamma[ch] + beta[ch];
    }
}

extern "C" void kernel_launch(void* const* d_in, const int* in_sizes, int n_in,
                              void* d_out, int out_size, void* d_ws, size_t ws_size,
                              hipStream_t stream) {
    const float* H      = (const float*)d_in[0];
    const float* E      = (const float*)d_in[1];
    const int*   ht     = (const int*)  d_in[2];
    const float* R      = (const float*)d_in[3];
    const float* W_fwd  = (const float*)d_in[4];
    const float* b_fwd  = (const float*)d_in[5];
    const float* W_back = (const float*)d_in[6];
    const float* b_back = (const float*)d_in[7];
    const float* gamma  = (const float*)d_in[8];
    const float* beta   = (const float*)d_in[9];
    float* out = (float*)d_out;

    const int n_nodes = in_sizes[0] / EMB;
    const int m       = in_sizes[2] / 2;

    // workspace layout
    char* ws = (char*)d_ws;
    const size_t wbytes = (size_t)EMB * K4D * sizeof(ushort_t);   // 128KB each
    ushort_t* Whi_f = (ushort_t*)(ws + 0 * wbytes);
    ushort_t* Wlo_f = (ushort_t*)(ws + 1 * wbytes);
    ushort_t* Whi_b = (ushort_t*)(ws + 2 * wbytes);
    ushort_t* Wlo_b = (ushort_t*)(ws + 3 * wbytes);
    float* cntf = (float*)(ws + 4 * wbytes);                      // cntf, cntb contiguous
    float* cntb = cntf + n_nodes;

    // 1) zero agg (lives in d_out) and both count arrays (contiguous)
    {
        long ztot = (long)n_nodes * EMB;
        int blocks = (int)((ztot + 255) / 256);
        if (blocks > 4096) blocks = 4096;
        mp_zero_kernel<<<blocks, 256, 0, stream>>>(out, ztot);
        mp_zero_kernel<<<(2 * n_nodes + 255) / 256, 256, 0, stream>>>(cntf, 2L * n_nodes);
    }
    // 2) weight split
    mp_wprep_kernel<<<(EMB * K4D + 255) / 256, 256, 0, stream>>>(
        W_fwd, W_back, Whi_f, Wlo_f, Whi_b, Wlo_b);
    // 3) per-direction counts
    mp_counts_kernel<<<(m + 255) / 256, 256, 0, stream>>>(ht, cntf, cntb, m);
    // 4) fused gather + WMMA GEMM + scatter
    mp_gemm_scatter_kernel<<<(m + 15) / 16, 128, 0, stream>>>(
        H, E, ht, R, Whi_f, Wlo_f, Whi_b, Wlo_b, out, m);
    // 5) per-node finalize (one wave32 per node)
    mp_finalize_kernel<<<(n_nodes + 7) / 8, 256, 0, stream>>>(
        out, H, cntf, cntb, b_fwd, b_back, gamma, beta, n_nodes);
}
